// RotaryAttention_26199300505726
// MI455X (gfx1250) — matrix operand order repaired and checked by reading, not compiled, not run on hardware
//
#include <hip/hip_runtime.h>

// ---------------------------------------------------------------------------
// Types for CDNA5 WMMA (wave32): v_wmma_f32_16x16x32_bf16
// ---------------------------------------------------------------------------
typedef __bf16 bf16_t;
typedef __attribute__((ext_vector_type(16))) __bf16 v16bf;
typedef __attribute__((ext_vector_type(8)))  float  v8f;
typedef __attribute__((ext_vector_type(4)))  unsigned int v4u;
typedef __attribute__((ext_vector_type(8)))  int v8i;
typedef __attribute__((ext_vector_type(4)))  int v4i;

union ABFrag { uint4 q[2]; v16bf f; };   // 32B: one 16-elem bf16 operand per lane
union CFrag  { v8f v; float f[8]; };     // 16x16 f32 accumulator slice per lane

__device__ inline v8f wmma_bf16(const v16bf& a, const v16bf& b, const v8f& c) {
  return __builtin_amdgcn_wmma_f32_16x16x32_bf16(false, a, false, b,
                                                 (short)0, c, false, false);
}

// ---------------------------------------------------------------------------
// TDM: issue a 2-D tensor tile load (128 rows x 32 bf16) into LDS.
// D# group0: count=1, lds_addr, global_addr(57b), type=2.
// D# group1: data_size=4B units, pad_enable: pad 4 dwords after every
//            16 dwords  ->  LDS rows of 40 bf16 (bank-conflict-free b128).
// Tensor dims in 4-byte units: row = 16 units (32 bf16), K-stride = K/2.
// ---------------------------------------------------------------------------
__device__ inline void tdm_load_tile(unsigned lds_off, const bf16_t* gptr,
                                     unsigned k_units, unsigned rows) {
  unsigned long long ga = (unsigned long long)(size_t)gptr;
  v4u g0;
  g0[0] = 1u;                                   // count=1 (valid descriptor)
  g0[1] = lds_off;                              // LDS byte address
  g0[2] = (unsigned)ga;                         // global_addr[31:0]
  g0[3] = (unsigned)((ga >> 32) & 0x01FFFFFFu)  // global_addr[56:32]
          | (2u << 30);                         // type=2 ("image")
  v8i g1;
  g1[0] = (int)((2u << 16)                      // data_size = 4 bytes
                | (1u << 20)                    // pad_enable
                | (3u << 22)                    // pad_interval: 16 dwords
                | (3u << 25));                  // pad_amount:   4 dwords
  g1[1] = (int)((k_units & 0xFFFFu) << 16);             // tensor_dim0[15:0]
  g1[2] = (int)(((k_units >> 16) & 0xFFFFu)             // tensor_dim0[31:16]
                | ((rows & 0xFFFFu) << 16));            // tensor_dim1[15:0]
  g1[3] = (int)(((rows >> 16) & 0xFFFFu)                // tensor_dim1[31:16]
                | (16u << 16));                         // tile_dim0 = 16 units
  g1[4] = (int)128u;                                    // tile_dim1 = 128 rows
  g1[5] = (int)k_units;                                 // dim0_stride[31:0]
  g1[6] = 0;                                            // stride bits, 0
  g1[7] = 0;
  v4i gz = {0, 0, 0, 0};                        // groups 2/3: 2-D tensor
#if __clang_major__ >= 23
  v8i gz8 = {0, 0, 0, 0, 0, 0, 0, 0};
  __builtin_amdgcn_tensor_load_to_lds(g0, g1, gz, gz, gz8, 0);
#else
  __builtin_amdgcn_tensor_load_to_lds(g0, g1, gz, gz, 0);
#endif
}

__device__ inline unsigned lds_byte_off(const void* p) {
  return (unsigned)(size_t)p;  // flat LDS aperture: low 32 bits are the LDS offset
}

// ---------------------------------------------------------------------------
// cast f32 -> bf16 (vectorized)
// ---------------------------------------------------------------------------
__global__ void cast_f32_bf16(const float4* __restrict__ in,
                              bf16_t* __restrict__ out, int n4) {
  int i = blockIdx.x * 256 + threadIdx.x;
  if (i >= n4) return;
  float4 v = in[i];
  size_t o = (size_t)i * 4;
  out[o + 0] = (bf16_t)v.x;
  out[o + 1] = (bf16_t)v.y;
  out[o + 2] = (bf16_t)v.z;
  out[o + 3] = (bf16_t)v.w;
}

// ---------------------------------------------------------------------------
// W [2048,2048] f32 row-major  ->  Wt [N,K] bf16 (transposed), LDS-tiled
// ---------------------------------------------------------------------------
__global__ void wtrans_kernel(const float* __restrict__ W,
                              bf16_t* __restrict__ Wt) {
  __shared__ float tile[32][33];
  const int bx = blockIdx.x * 32, by = blockIdx.y * 32;
  const int tx = threadIdx.x, ty = threadIdx.y;
#pragma unroll
  for (int j = 0; j < 32; j += 8)
    tile[ty + j][tx] = W[(size_t)(by + ty + j) * 2048 + bx + tx];
  __syncthreads();
#pragma unroll
  for (int j = 0; j < 32; j += 8)
    Wt[(size_t)(bx + ty + j) * 2048 + by + tx] = (bf16_t)tile[tx][ty + j];
}

// ---------------------------------------------------------------------------
// GEMM: C[M,N] = A_bf16[M,K] * Bt_bf16[N,K]^T + bias
//   MODE 0: bf16 row-major [M,N]
//   MODE 1: f32  row-major [M,N]   (final output)
//   MODE 2: bf16 Vt layout  [b,h,hd,s]  (per-head transpose for P*V)
// 128x128 block tile, 8 waves (4 M x 2 N), wave tile 32x64 = 2x4 WMMAs.
// K-slabs of 32 staged by the Tensor Data Mover into double-buffered LDS;
// wave 0 issues the DMA + s_wait_tensorcnt, barriers release the others.
// ---------------------------------------------------------------------------
template <int MODE>
__global__ __launch_bounds__(256)
void gemm_bf16_kernel(const bf16_t* __restrict__ A,
                      const bf16_t* __restrict__ Bt,
                      const float* __restrict__ bias,
                      void* __restrict__ out, int M, int N, int K) {
  __shared__ __align__(16) bf16_t As[2][128][40];  // +8 bf16 pad via TDM padding
  __shared__ __align__(16) bf16_t Bs[2][128][40];

  const int tid  = threadIdx.x;
  const int lane = tid & 31, wave = tid >> 5;
  const int wm = wave & 3, wn = wave >> 2;
  const int l = lane & 15, hi = lane >> 4;
  const int m0 = blockIdx.y * 128, n0 = blockIdx.x * 128;

  const bf16_t* gA = A  + (size_t)m0 * K;  // tile row origin, advance by k
  const bf16_t* gB = Bt + (size_t)n0 * K;
  const unsigned kU = (unsigned)(K >> 1);  // K in 4-byte units

  CFrag acc[2][4];
#pragma unroll
  for (int i = 0; i < 2; i++)
#pragma unroll
    for (int j = 0; j < 4; j++) {
      v8f z = {0.f, 0.f, 0.f, 0.f, 0.f, 0.f, 0.f, 0.f};
      acc[i][j].v = z;
    }

  const int NK = K >> 5;  // K-slabs of 32
  if (wave == 0) {        // prologue: DMA slab 0 into buffer 0
    tdm_load_tile(lds_byte_off(&As[0][0][0]), gA, kU, (unsigned)M);
    tdm_load_tile(lds_byte_off(&Bs[0][0][0]), gB, kU, (unsigned)N);
  }

  for (int i = 0; i < NK; ++i) {
    const int cur = i & 1;
    if (wave == 0) {
      if (i + 1 < NK) {  // DMA next slab into the other buffer, then wait
        const int nxt = (i + 1) & 1;
        tdm_load_tile(lds_byte_off(&As[nxt][0][0]), gA + (i + 1) * 32, kU,
                      (unsigned)M);
        tdm_load_tile(lds_byte_off(&Bs[nxt][0][0]), gB + (i + 1) * 32, kU,
                      (unsigned)N);
        __builtin_amdgcn_s_wait_tensorcnt(2);  // in-order: slab i complete
      } else {
        __builtin_amdgcn_s_wait_tensorcnt(0);
      }
    }
    __syncthreads();  // slab i visible to all waves

    // A fragments (ISA 16-bit A layout: lanes 0-15 K[0:8)+[16:24),
    // lanes 16-31 K[8:16)+[24:32), row M = lane&15)
    ABFrag fa[2];
#pragma unroll
    for (int mt = 0; mt < 2; mt++) {
      const bf16_t* p = &As[cur][32 * wm + 16 * mt + l][hi * 8];
      fa[mt].q[0] = *(const uint4*)p;
      fa[mt].q[1] = *(const uint4*)(p + 16);
    }
#pragma unroll
    for (int nt = 0; nt < 4; nt++) {
      // B fragment: lane n holds column N=n, K[0:8)+[16:24) (lanes 0-15) / K[8:16)+[24:32)
      ABFrag fb;
      const bf16_t* p = &Bs[cur][64 * wn + 16 * nt + l][hi * 8];
      fb.q[0] = *(const uint4*)p;
      fb.q[1] = *(const uint4*)(p + 16);
#pragma unroll
      for (int mt = 0; mt < 2; mt++)
        acc[mt][nt].v = wmma_bf16(fa[mt].f, fb.f, acc[mt][nt].v);
    }
    __syncthreads();  // done reading slab i before TDM overwrites this buffer
  }

  // epilogue — C layout: VGPR v, lanes 0-15: M=v; lanes 16-31: M=8+v; N=lane&15
#pragma unroll
  for (int mt = 0; mt < 2; mt++)
#pragma unroll
    for (int nt = 0; nt < 4; nt++) {
      const int col = n0 + 64 * wn + 16 * nt + l;
      const int rb  = m0 + 32 * wm + 16 * mt + hi * 8;
      const float bv = bias[col];
#pragma unroll
      for (int v = 0; v < 8; v++) {
        const int row = rb + v;
        const float val = acc[mt][nt].f[v] + bv;
        if (MODE == 0) {
          ((bf16_t*)out)[(size_t)row * N + col] = (bf16_t)val;
        } else if (MODE == 1) {
          ((float*)out)[(size_t)row * N + col] = val;
        } else {  // Vt: [b, h, hd, s]
          const int b = row >> 7, s = row & 127, h = col >> 7, d = col & 127;
          ((bf16_t*)out)[((size_t)(b * 16 + h) * 128 + d) * 128 + s] =
              (bf16_t)val;
        }
      }
    }
}

// ---------------------------------------------------------------------------
// RoPE: lin[token, 2048] bf16 -> out[b, h, s, hd] bf16 (interleaved pairs)
// ---------------------------------------------------------------------------
__global__ void rope_kernel(const bf16_t* __restrict__ lin,
                            bf16_t* __restrict__ out) {
  const int idx = blockIdx.x * 256 + threadIdx.x;  // exactly 8192*1024 threads
  const int token = idx >> 10;
  const int p = idx & 1023;
  const int h = p >> 6, i = p & 63;
  const int b = token >> 7, s = token & 127;
  const float inv = powf(10000.0f, -(float)(2 * i) * (1.0f / 128.0f));
  float sn, cs;
  sincosf((float)s * inv, &sn, &cs);
  const size_t src = (size_t)token * 2048 + h * 128 + 2 * i;
  const float e = (float)lin[src], o = (float)lin[src + 1];
  const size_t dst = ((size_t)(b * 16 + h) * 128 + s) * 128 + 2 * i;
  out[dst]     = (bf16_t)(e * cs - o * sn);
  out[dst + 1] = (bf16_t)(o * cs + e * sn);
}

// ---------------------------------------------------------------------------
// Attention core: one block per (b,h). 8 waves x 16 query rows.
// S = Q*K^T (WMMA), causal-masked softmax (shfl over 16-lane halves),
// P round-trips through per-wave LDS (C-layout -> A-layout), O = P*V (WMMA).
// ---------------------------------------------------------------------------
__global__ __launch_bounds__(256)
void attn_kernel(const bf16_t* __restrict__ Q, const bf16_t* __restrict__ Kr,
                 const bf16_t* __restrict__ Vt, bf16_t* __restrict__ ctx) {
  __shared__ __align__(16) bf16_t Psh[8][16][136];  // per-wave P tile (padded)

  const int bh = blockIdx.x;
  const int b = bh >> 4, h = bh & 15;
  const int tid = threadIdx.x, lane = tid & 31, wave = tid >> 5;
  const int l = lane & 15, hi = lane >> 4;
  const int q0 = wave * 16;

  const bf16_t* Qb = Q  + (size_t)bh * 16384;  // [s, hd]
  const bf16_t* Kb = Kr + (size_t)bh * 16384;  // [s, hd]
  const bf16_t* Vb = Vt + (size_t)bh * 16384;  // [hd, s]

  // ---- S = Q * K^T : 8 N-tiles (128 keys), K-dim = 128 = 4 x 32 ----
  CFrag sacc[8];
#pragma unroll
  for (int nb = 0; nb < 8; nb++) {
    v8f z = {0.f, 0.f, 0.f, 0.f, 0.f, 0.f, 0.f, 0.f};
    sacc[nb].v = z;
  }
#pragma unroll
  for (int ks = 0; ks < 4; ks++) {
    ABFrag fa;
    const bf16_t* pq = Qb + (size_t)(q0 + l) * 128 + ks * 32 + hi * 8;
    fa.q[0] = *(const uint4*)pq;
    fa.q[1] = *(const uint4*)(pq + 16);
#pragma unroll
    for (int nb = 0; nb < 8; nb++) {
      ABFrag fb;  // B column n = key row (16*nb + n), contiguous over hd
      const bf16_t* pk = Kb + (size_t)(16 * nb + l) * 128 + ks * 32 + hi * 8;
      fb.q[0] = *(const uint4*)pk;
      fb.q[1] = *(const uint4*)(pk + 16);
      sacc[nb].v = wmma_bf16(fa.f, fb.f, sacc[nb].v);
    }
  }

  // ---- causal softmax over 128 keys; rows live in 16-lane halves ----
  const float scale = 0.08838834764831845f;  // 1/sqrt(128)
#pragma unroll
  for (int v = 0; v < 8; v++) {
    const int qrow = q0 + hi * 8 + v;
    float lv[8];
    float m = -3.0e38f;
#pragma unroll
    for (int nb = 0; nb < 8; nb++) {
      const int col = 16 * nb + l;
      float t = sacc[nb].f[v] * scale;
      if (col > qrow) t = -1.0e9f;
      lv[nb] = t;
      m = fmaxf(m, t);
    }
    m = fmaxf(m, __shfl_xor(m, 1, 32));
    m = fmaxf(m, __shfl_xor(m, 2, 32));
    m = fmaxf(m, __shfl_xor(m, 4, 32));
    m = fmaxf(m, __shfl_xor(m, 8, 32));
    float pe[8];
    float sum = 0.f;
#pragma unroll
    for (int nb = 0; nb < 8; nb++) {
      pe[nb] = __expf(lv[nb] - m);
      sum += pe[nb];
    }
    sum += __shfl_xor(sum, 1, 32);
    sum += __shfl_xor(sum, 2, 32);
    sum += __shfl_xor(sum, 4, 32);
    sum += __shfl_xor(sum, 8, 32);
    const float inv = 1.0f / sum;
#pragma unroll
    for (int nb = 0; nb < 8; nb++)
      Psh[wave][hi * 8 + v][16 * nb + l] = (bf16_t)(pe[nb] * inv);
  }

  // ---- O = P * V : 8 hd-tiles, K-dim = 128 keys = 4 x 32 ----
  CFrag oacc[8];
#pragma unroll
  for (int nb = 0; nb < 8; nb++) {
    v8f z = {0.f, 0.f, 0.f, 0.f, 0.f, 0.f, 0.f, 0.f};
    oacc[nb].v = z;
  }
#pragma unroll
  for (int ks = 0; ks < 4; ks++) {
    ABFrag fa;
    const bf16_t* pp = &Psh[wave][l][ks * 32 + hi * 8];
    fa.q[0] = *(const uint4*)pp;
    fa.q[1] = *(const uint4*)(pp + 16);
#pragma unroll
    for (int nb = 0; nb < 8; nb++) {
      ABFrag fb;  // B column n = Vt row (16*nb + n), contiguous over keys
      const bf16_t* pv = Vb + (size_t)(16 * nb + l) * 128 + ks * 32 + hi * 8;
      fb.q[0] = *(const uint4*)pv;
      fb.q[1] = *(const uint4*)(pv + 16);
      oacc[nb].v = wmma_bf16(fa.f, fb.f, oacc[nb].v);
    }
  }

  // ---- write ctx[token, 2048] bf16 ----
#pragma unroll
  for (int nb = 0; nb < 8; nb++)
#pragma unroll
    for (int v = 0; v < 8; v++) {
      const int token = b * 128 + q0 + hi * 8 + v;
      ctx[(size_t)token * 2048 + h * 128 + 16 * nb + l] = (bf16_t)oacc[nb].f[v];
    }
}

// ---------------------------------------------------------------------------
// launch
// ---------------------------------------------------------------------------
extern "C" void kernel_launch(void* const* d_in, const int* in_sizes, int n_in,
                              void* d_out, int out_size, void* d_ws,
                              size_t ws_size, hipStream_t stream) {
  (void)in_sizes; (void)n_in; (void)out_size; (void)ws_size;
  const float* x  = (const float*)d_in[0];
  const float* Wq = (const float*)d_in[1];
  const float* bq = (const float*)d_in[2];
  const float* Wk = (const float*)d_in[3];
  const float* bk = (const float*)d_in[4];
  const float* Wv = (const float*)d_in[5];
  const float* bv = (const float*)d_in[6];
  const float* Wo = (const float*)d_in[7];
  const float* bo = (const float*)d_in[8];
  float* out = (float*)d_out;

  const size_t SZ_X = (size_t)8192 * 2048 * 2;  // 32 MB bf16 activations
  const size_t SZ_W = (size_t)2048 * 2048 * 2;  //  8 MB bf16 weights
  char* w = (char*)d_ws;
  bf16_t* xb   = (bf16_t*)w; w += SZ_X;
  bf16_t* WqT  = (bf16_t*)w; w += SZ_W;
  bf16_t* WkT  = (bf16_t*)w; w += SZ_W;
  bf16_t* WvT  = (bf16_t*)w; w += SZ_W;
  bf16_t* WoT  = (bf16_t*)w; w += SZ_W;
  bf16_t* Qlin = (bf16_t*)w; w += SZ_X;
  bf16_t* Klin = (bf16_t*)w; w += SZ_X;
  bf16_t* Vt   = (bf16_t*)w; w += SZ_X;
  bf16_t* Qr   = (bf16_t*)w; w += SZ_X;
  bf16_t* Kr   = (bf16_t*)w; w += SZ_X;
  bf16_t* ctx  = Qlin;  // Qlin dead after RoPE; reuse for attention output

  cast_f32_bf16<<<16384, 256, 0, stream>>>((const float4*)x, xb, 4194304);
  dim3 tg(64, 64), tb(32, 8);
  wtrans_kernel<<<tg, tb, 0, stream>>>(Wq, WqT);
  wtrans_kernel<<<tg, tb, 0, stream>>>(Wk, WkT);
  wtrans_kernel<<<tg, tb, 0, stream>>>(Wv, WvT);
  wtrans_kernel<<<tg, tb, 0, stream>>>(Wo, WoT);

  dim3 gg(16, 64);  // N/128, M/128
  gemm_bf16_kernel<0><<<gg, 256, 0, stream>>>(xb, WqT, bq, Qlin, 8192, 2048, 2048);
  gemm_bf16_kernel<0><<<gg, 256, 0, stream>>>(xb, WkT, bk, Klin, 8192, 2048, 2048);
  gemm_bf16_kernel<2><<<gg, 256, 0, stream>>>(xb, WvT, bv, Vt,   8192, 2048, 2048);

  rope_kernel<<<32768, 256, 0, stream>>>(Qlin, Qr);
  rope_kernel<<<32768, 256, 0, stream>>>(Klin, Kr);

  attn_kernel<<<1024, 256, 0, stream>>>(Qr, Kr, Vt, ctx);

  gemm_bf16_kernel<1><<<gg, 256, 0, stream>>>(ctx, WoT, bo, out, 8192, 2048, 2048);
}
